// OptimalTransportLoss_63196148793590
// MI455X (gfx1250) — matrix operand order repaired and checked by reading, not compiled, not run on hardware
//
#include <hip/hip_runtime.h>

typedef __attribute__((ext_vector_type(16))) __bf16 v16bf;
typedef __attribute__((ext_vector_type(8)))  float  v8f;

static constexpr int   GN    = 64;          // pooled grid side
static constexpr int   GNN   = GN * GN;     // 4096
static constexpr int   POOL  = 8;
static constexpr int   HW    = 512;
static constexpr float REGP  = 0.1f;
static constexpr int   ITERS = 50;
static constexpr float EPSF  = 1e-8f;

// ---------------------------------------------------------------------------
// Kernel 1: pooled window-sum (avg_pool * d^2 == window sum) + normalization.
// One block per (batch, image) pair; 256 threads, 16 cells each.
// ---------------------------------------------------------------------------
__global__ __launch_bounds__(256) void ot_pool_norm(
    const float* __restrict__ pred, const float* __restrict__ gt,
    float* __restrict__ norm, float* __restrict__ sums) {
  const int img = blockIdx.x;              // 2*b + which
  const int b   = img >> 1;
  const float* src = ((img & 1) ? gt : pred) + (size_t)b * HW * HW;

  float r[16];
  float local = 0.f;
#pragma unroll
  for (int i = 0; i < 16; ++i) {
    const int c  = threadIdx.x + i * 256;  // pooled cell 0..4095
    const int oy = c >> 6, ox = c & 63;
    const float* w = src + (oy * POOL) * HW + ox * POOL;
    float s = 0.f;
#pragma unroll
    for (int dy = 0; dy < POOL; ++dy)
#pragma unroll
      for (int dx = 0; dx < POOL; ++dx) s += w[dy * HW + dx];
    r[i] = s;
    local += s;
  }

  __shared__ float red[256];
  red[threadIdx.x] = local;
  __syncthreads();
  for (int off = 128; off > 0; off >>= 1) {
    if (threadIdx.x < off) red[threadIdx.x] += red[threadIdx.x + off];
    __syncthreads();
  }
  const float total = red[0];
  const float scl   = 1.f / fmaxf(total, 1e-8f);
  float* dst = norm + (size_t)img * GNN;
#pragma unroll
  for (int i = 0; i < 16; ++i) dst[threadIdx.x + i * 256] = r[i] * scl;
  if (threadIdx.x == 0) sums[img] = total;   // unclipped, for validity test
}

// ---------------------------------------------------------------------------
// WMMA fragment loaders, bf16 LDS operands only (wave32 layouts, ISA 7.12.2).
// A element e -> k = kb + 16*g + e                  (row = tile_m + lane&15)
//   row-major bf16: one contiguous 16-element (32 B) run.
// B element e -> k = kb + 16*g + e                  (col = tile_n + lane&15)
//   from a COLUMN-ACCESSED matrix Xc where Xc[n*GN + k] = B[k][n]
//   (true for symmetric K/KC stored row-major, and for U/V stored transposed)
//   -> one contiguous 16-element (32 B) run.
// ---------------------------------------------------------------------------
__device__ __forceinline__ v16bf load_a_bf(const __bf16* X, int mrow, int kb, int g) {
  v16bf a;
  const __bf16* p = X + mrow * GN + kb + 16 * g;
#pragma unroll
  for (int e = 0; e < 16; ++e) a[e] = p[e];
  return a;
}
__device__ __forceinline__ v16bf load_b_col(const __bf16* Xc, int kb, int ncol, int g) {
  v16bf bb;
  const __bf16* p = Xc + ncol * GN + kb + 16 * g;
#pragma unroll
  for (int e = 0; e < 16; ++e) bb[e] = p[e];
  return bb;
}

// Z = A(64x64 bf16 row-major) @ B(64x64 bf16 column-accessed)
// 16 waves, one 16x16 output tile each; D stored bf16 or f32.
template <bool OUT_BF>
__device__ __forceinline__ void gemm64(const __bf16* Amat, const __bf16* Bcol,
                                       __bf16* Zb, float* Zf, int wave, int lane) {
  const int tm = (wave >> 2) * 16, tn = (wave & 3) * 16;
  const int g = lane >> 4, l = lane & 15;
  v8f acc = {};
#pragma unroll
  for (int kb = 0; kb < GN; kb += 32) {
    v16bf a  = load_a_bf(Amat, tm + l, kb, g);
    v16bf bb = load_b_col(Bcol, kb, tn + l, g);
    acc = __builtin_amdgcn_wmma_f32_16x16x32_bf16(false, a, false, bb,
                                                  (short)0, acc, false, false);
  }
  const int zo = (tm + 8 * g) * GN + tn + l;
  if (OUT_BF) {
#pragma unroll
    for (int r = 0; r < 8; ++r) Zb[zo + r * GN] = (__bf16)acc[r];
  } else {
#pragma unroll
    for (int r = 0; r < 8; ++r) Zf[zo + r * GN] = acc[r];
  }
}

// ---------------------------------------------------------------------------
// Kernel 2: full Sinkhorn per batch in LDS.  K = Ky = Kx (H==W), symmetric.
// K@v on image V  ==  K · V · K   (two 64x64x64 bf16 WMMA GEMMs)
// M@v             ==  KC·V·K + K·V·KC   (KC = K ∘ C_1d)
// All WMMA operands live in LDS as bf16; f32 kept only where precision is
// consumed (division denominators, final dot product).
// ---------------------------------------------------------------------------
__global__ __launch_bounds__(512) void ot_sinkhorn(
    const float* __restrict__ norm, const float* __restrict__ sums,
    float* __restrict__ out) {
  __shared__ __bf16 Kb[GNN], KCb[GNN];       // Gibbs kernel / kernel*cost, 8 KB each
  __shared__ __bf16 Ut[GNN], Vt[GNN];        // u, v images, TRANSPOSED bf16
  __shared__ __bf16 Tb[GNN];                 // intermediate  K·X  (bf16)
  __shared__ float  Uf[GNN];                 // u in f32 (final dot product)
  __shared__ float  Af[GNN], Bf[GNN];        // normalized marginals a, b
  __shared__ float  Sf[GNN];                 // K·X·K in f32 (denominator / cost)

  const int b = blockIdx.x;
  const int t = threadIdx.x;
  const int wave = t >> 5, lane = t & 31;

  const float* an = norm + (size_t)(2 * b) * GNN;
  const float* bn = norm + (size_t)(2 * b + 1) * GNN;

  for (int e = t; e < GNN; e += 512) {
    const int i = e >> 6, j = e & 63;
    const float d = (float)(i - j);
    const float c = d * d;
    const float k = __expf(-c / REGP);
    Kb[e]  = (__bf16)k;
    KCb[e] = (__bf16)(k * c);
    Af[e]  = an[e];
    Bf[e]  = bn[e];
    Uf[e]  = 1.f;                // scan init: u = ones (transpose-invariant)
    Ut[e]  = (__bf16)1.f;
    Vt[e]  = (__bf16)0.f;
  }
  __syncthreads();

  for (int it = 0; it < ITERS; ++it) {
    // v = b ./ (K^T u + eps) ; K symmetric -> Ky U Kx
    gemm64<true >(Kb, Ut, Tb, nullptr, wave, lane); __syncthreads();
    gemm64<false>(Tb, Kb, nullptr, Sf, wave, lane); __syncthreads();
    for (int e = t; e < GNN; e += 512) {
      const float v = Bf[e] / (Sf[e] + EPSF);
      Vt[(e & 63) * GN + (e >> 6)] = (__bf16)v;          // store transposed
    }
    __syncthreads();
    // u = a ./ (K v + eps)
    gemm64<true >(Kb, Vt, Tb, nullptr, wave, lane); __syncthreads();
    gemm64<false>(Tb, Kb, nullptr, Sf, wave, lane); __syncthreads();
    for (int e = t; e < GNN; e += 512) {
      const float u = Af[e] / (Sf[e] + EPSF);
      Ut[(e & 63) * GN + (e >> 6)] = (__bf16)u;          // store transposed
      Uf[e] = u;
    }
    __syncthreads();
  }

  // cost = u^T (M v) = sum U ∘ (KC·V·K + K·V·KC)
  float cost = 0.f;
  gemm64<true >(KCb, Vt, Tb, nullptr, wave, lane); __syncthreads();
  gemm64<false>(Tb, Kb, nullptr, Sf, wave, lane);  __syncthreads();
  for (int e = t; e < GNN; e += 512) cost += Uf[e] * Sf[e];
  __syncthreads();
  gemm64<true >(Kb, Vt, Tb, nullptr, wave, lane);  __syncthreads();
  gemm64<false>(Tb, KCb, nullptr, Sf, wave, lane); __syncthreads();
  for (int e = t; e < GNN; e += 512) cost += Uf[e] * Sf[e];
  __syncthreads();

  // block reduction (reuse Sf as scratch)
  Sf[t] = cost;
  __syncthreads();
  for (int off = 256; off > 0; off >>= 1) {
    if (t < off) Sf[t] += Sf[t + off];
    __syncthreads();
  }
  if (t == 0) {
    const float asum = sums[2 * b], bsum = sums[2 * b + 1];
    const float v = (asum > 0.5f && bsum > 0.5f) ? Sf[0] : 0.f;
    atomicAdd(out, v * (1.f / 16.f));
  }
}

// ---------------------------------------------------------------------------
extern "C" void kernel_launch(void* const* d_in, const int* in_sizes, int n_in,
                              void* d_out, int out_size, void* d_ws, size_t ws_size,
                              hipStream_t stream) {
  (void)n_in; (void)out_size; (void)ws_size;
  const float* pred = (const float*)d_in[0];
  const float* gt   = (const float*)d_in[1];
  const int B = in_sizes[0] / (HW * HW);     // 16

  float* norm = (float*)d_ws;                // [2B][4096]
  float* sums = norm + (size_t)(2 * B) * GNN;
  float* out  = (float*)d_out;

  hipMemsetAsync(out, 0, sizeof(float), stream);
  ot_pool_norm<<<2 * B, 256, 0, stream>>>(pred, gt, norm, sums);
  ot_sinkhorn<<<B, 512, 0, stream>>>(norm, sums, out);
}
